// CustomLSTMCell_18159121727814
// MI455X (gfx1250) — matrix operand order repaired and checked by reading, not compiled, not run on hardware
//
#include <hip/hip_runtime.h>

typedef __bf16 v16bf __attribute__((ext_vector_type(16)));
typedef float  v8f   __attribute__((ext_vector_type(8)));

#define BSZ  8192
#define HN   1024
#define KTOT 2048
#define NKS  64            // K-steps of 32
#define NMT  512           // 16-row M tiles
#define NNT  64            // 16-col N tiles (per gate)

// Hardware tanh (V_TANH_F32) if available; sigmoid via tanh identity.
#if __has_builtin(__builtin_amdgcn_tanhf)
__device__ __forceinline__ float fast_tanh(float x) { return __builtin_amdgcn_tanhf(x); }
#else
__device__ __forceinline__ float fast_tanh(float x) { return tanhf(x); }
#endif
__device__ __forceinline__ float fast_sigmoid(float x) {
    return __builtin_fmaf(0.5f, fast_tanh(0.5f * x), 0.5f);
}

// ---------------------------------------------------------------------------
// Pass 1: pack A = [x | h_prev] (f32) into fragment-major bf16 blocks in ws.
// Block (mt, ks): 32 lanes x 16 bf16 (1 KB), lane layout per ISA A-striping:
//   lane<16 : row = mt*16+lane,    K = ks*32 + {0..7, 16..23}
//   lane>=16: row = mt*16+lane-16, K = ks*32 + {8..15, 24..31}
// ---------------------------------------------------------------------------
__global__ __launch_bounds__(256)
void pack_A(const float* __restrict__ x, const float* __restrict__ hprev,
            __bf16* __restrict__ wsA)
{
    const int lane = threadIdx.x & 31;
    const int wid  = blockIdx.x * 8 + (threadIdx.x >> 5);   // 0..32767
    const int mt   = wid >> 6;                              // 0..511
    const int ks   = wid & 63;                              // 0..63

    const long row  = (long)mt * 16 + (lane & 15);
    const int  koff = (lane >> 4) * 8;
    const int  kA   = ks * 32 + koff;        // chunk of 8 contiguous K
    const int  kB   = kA + 16;               // chunk of 8 contiguous K
    const float* srcA = (kA < HN) ? (x + row * HN + kA)
                                  : (hprev + row * HN + (kA - HN));
    const float* srcB = (kB < HN) ? (x + row * HN + kB)
                                  : (hprev + row * HN + (kB - HN));
    float4 fa0 = ((const float4*)srcA)[0];
    float4 fa1 = ((const float4*)srcA)[1];
    float4 fb0 = ((const float4*)srcB)[0];
    float4 fb1 = ((const float4*)srcB)[1];
    v16bf v;
    v[0]=(__bf16)fa0.x;  v[1]=(__bf16)fa0.y;  v[2]=(__bf16)fa0.z;  v[3]=(__bf16)fa0.w;
    v[4]=(__bf16)fa1.x;  v[5]=(__bf16)fa1.y;  v[6]=(__bf16)fa1.z;  v[7]=(__bf16)fa1.w;
    v[8]=(__bf16)fb0.x;  v[9]=(__bf16)fb0.y;  v[10]=(__bf16)fb0.z; v[11]=(__bf16)fb0.w;
    v[12]=(__bf16)fb1.x; v[13]=(__bf16)fb1.y; v[14]=(__bf16)fb1.z; v[15]=(__bf16)fb1.w;
    *(v16bf*)(wsA + (long)wid * 512 + lane * 16) = v;
}

// ---------------------------------------------------------------------------
// Pass 2: pack the 4 gate weight matrices into fragment-major bf16 blocks.
// Block (g, nt, ks): lane layout per ISA B-striping:
//   lane<16 : col = nt*16+lane,    K = ks*32 + {0..7, 16..23}
//   lane>=16: col = nt*16+lane-16, K = ks*32 + {8..15, 24..31}
// W is [K, HN] row-major, so the per-lane K run is a stride-HN gather (once).
// ---------------------------------------------------------------------------
__global__ __launch_bounds__(256)
void pack_B(const float* __restrict__ Wf, const float* __restrict__ Wi,
            const float* __restrict__ Wo, const float* __restrict__ Wc,
            __bf16* __restrict__ wsB)
{
    const int lane = threadIdx.x & 31;
    const int wid  = blockIdx.x * 8 + (threadIdx.x >> 5);   // 0..16383
    const int g    = wid >> 12;                             // 0..3  (f,i,o,C)
    const int nt   = (wid >> 6) & 63;                       // 0..63
    const int ks   = wid & 63;                              // 0..63
    const float* Wg = (g == 0) ? Wf : (g == 1) ? Wi : (g == 2) ? Wo : Wc;

    const int  n  = nt * 16 + (lane & 15);
    const long kb = (long)ks * 32 + (lane >> 4) * 8;
    v16bf v;
    #pragma unroll
    for (int j = 0; j < 16; ++j) v[j] = (__bf16)Wg[(kb + j + (j >= 8 ? 8 : 0)) * HN + n];
    *(v16bf*)(wsB + (long)wid * 512 + lane * 16) = v;
}

// ---------------------------------------------------------------------------
// GEMM + fused LSTM epilogue. No LDS, no barriers, no converts in the hot
// loop. Fully unrolled K loop: fragment offsets are compile-time immediates,
// software-pipeline copies are SSA renames (no v_mov, no hazard v_nops).
// Per wave: 32 rows x 16 cols x 4 gates = 8 WMMAs per K-step.
// ---------------------------------------------------------------------------
#define WMMA_BF16(A, B, C) \
    __builtin_amdgcn_wmma_f32_16x16x32_bf16(false, (A), false, (B), (short)0, (C), false, false)

__global__ __launch_bounds__(256)
void lstm_gemm_wmma(const __bf16* __restrict__ wsA, const __bf16* __restrict__ wsB,
                    const float* __restrict__ Cprev,
                    const float* __restrict__ bf_, const float* __restrict__ bi_,
                    const float* __restrict__ bo_, const float* __restrict__ bc_,
                    float* __restrict__ out)
{
    const int lane = threadIdx.x & 31;
    const int wave = threadIdx.x >> 5;
    const int wm   = wave >> 1;                 // 0..3
    const int wn   = wave & 1;                  // 0..1
    const int mt0  = blockIdx.y * 8 + wm * 2;   // two 16-row tiles
    const int mt1  = mt0 + 1;
    const int nt   = blockIdx.x * 2 + wn;       // one 16-col tile

    const __bf16* pa0 = wsA + (long)mt0 * NKS * 512 + lane * 16;
    const __bf16* pa1 = wsA + (long)mt1 * NKS * 512 + lane * 16;
    const __bf16* pb0 = wsB + ((long)(0 * NNT + nt) * NKS) * 512 + lane * 16;
    const __bf16* pb1 = wsB + ((long)(1 * NNT + nt) * NKS) * 512 + lane * 16;
    const __bf16* pb2 = wsB + ((long)(2 * NNT + nt) * NKS) * 512 + lane * 16;
    const __bf16* pb3 = wsB + ((long)(3 * NNT + nt) * NKS) * 512 + lane * 16;

    v8f dF0 = {}, dF1 = {}, dI0 = {}, dI1 = {};
    v8f dO0 = {}, dO1 = {}, dC0 = {}, dC1 = {};

    // prologue fragment loads (ks = 0)
    v16bf a0 = *(const v16bf*)pa0;
    v16bf a1 = *(const v16bf*)pa1;
    v16bf b0 = *(const v16bf*)pb0;
    v16bf b1 = *(const v16bf*)pb1;
    v16bf b2 = *(const v16bf*)pb2;
    v16bf b3 = *(const v16bf*)pb3;

    #pragma unroll
    for (int ks = 0; ks < NKS; ++ks) {
        v16bf a0n, a1n, b0n, b1n, b2n, b3n;
        if (ks + 1 < NKS) {                 // compile-time per unrolled copy
            const long off = (long)(ks + 1) * 512;  // folds into imm offset
            a0n = *(const v16bf*)(pa0 + off);
            a1n = *(const v16bf*)(pa1 + off);
            b0n = *(const v16bf*)(pb0 + off);
            b1n = *(const v16bf*)(pb1 + off);
            b2n = *(const v16bf*)(pb2 + off);
            b3n = *(const v16bf*)(pb3 + off);
        }
        dF0 = WMMA_BF16(a0, b0, dF0);  dF1 = WMMA_BF16(a1, b0, dF1);
        dI0 = WMMA_BF16(a0, b1, dI0);  dI1 = WMMA_BF16(a1, b1, dI1);
        dO0 = WMMA_BF16(a0, b2, dO0);  dO1 = WMMA_BF16(a1, b2, dO1);
        dC0 = WMMA_BF16(a0, b3, dC0);  dC1 = WMMA_BF16(a1, b3, dC1);
        if (ks + 1 < NKS) {                 // SSA rename after full unroll
            a0 = a0n; a1 = a1n;
            b0 = b0n; b1 = b1n; b2 = b2n; b3 = b3n;
        }
    }

    // ---- fused LSTM epilogue (hardware tanh; sigmoid via tanh identity) ----
    // C/D layout: VGPR i -> M = i + 8*(lane>=16), N = lane&15
    const int   gcol = nt * 16 + (lane & 15);
    const float bfv = bf_[gcol], biv = bi_[gcol], bov = bo_[gcol], bcv = bc_[gcol];

    #pragma unroll
    for (int tt = 0; tt < 2; ++tt) {
        const v8f gF = tt ? dF1 : dF0;
        const v8f gI = tt ? dI1 : dI0;
        const v8f gO = tt ? dO1 : dO0;
        const v8f gC = tt ? dC1 : dC0;
        const long rowb = (long)(tt ? mt1 : mt0) * 16 + ((lane >> 4) * 8);
        #pragma unroll
        for (int i2 = 0; i2 < 8; ++i2) {
            const long grow = rowb + i2;
            const float fgate = fast_sigmoid(gF[i2] + bfv);
            const float igate = fast_sigmoid(gI[i2] + biv);
            const float ogate = fast_sigmoid(gO[i2] + bov);
            const float ctil  = fast_tanh(gC[i2] + bcv);
            const float cp    = Cprev[grow * HN + gcol];
            const float ct    = fgate * cp + igate * ctil;
            const float ht    = ogate * fast_tanh(ct);
            out[grow * HN + gcol]                  = ht;   // h_t
            out[(long)BSZ * HN + grow * HN + gcol] = ct;   // C_t
        }
    }
}

extern "C" void kernel_launch(void* const* d_in, const int* in_sizes, int n_in,
                              void* d_out, int out_size, void* d_ws, size_t ws_size,
                              hipStream_t stream) {
    const float* x     = (const float*)d_in[0];
    const float* hprev = (const float*)d_in[1];
    const float* Cprev = (const float*)d_in[2];
    const float* Wf    = (const float*)d_in[3];
    const float* bf    = (const float*)d_in[4];
    const float* Wi    = (const float*)d_in[5];
    const float* bi    = (const float*)d_in[6];
    const float* Wc    = (const float*)d_in[7];
    const float* bc    = (const float*)d_in[8];
    const float* Wo    = (const float*)d_in[9];
    const float* bo    = (const float*)d_in[10];
    float* out = (float*)d_out;

    __bf16* wsA = (__bf16*)d_ws;                                   // 33.5 MB
    __bf16* wsB = (__bf16*)((char*)d_ws + (size_t)BSZ * KTOT * 2); // 16.8 MB

    // Pass 1: 512 mt * 64 ks waves / 8 waves per block
    pack_A<<<4096, 256, 0, stream>>>(x, hprev, wsA);
    // Pass 2: 4 g * 64 nt * 64 ks waves / 8 waves per block
    pack_B<<<2048, 256, 0, stream>>>(Wf, Wi, Wo, Wc, wsB);
    // GEMM: grid (N-blocks fastest so A panels are shared in L2 by adjacent WGs)
    lstm_gemm_wmma<<<dim3(HN / 32, BSZ / 128), 256, 0, stream>>>(
        wsA, wsB, Cprev, bf, bi, bo, bc, out);
}
